// GSC_YOLO_34986803593863
// MI455X (gfx1250) — matrix operand order repaired and checked by reading, not compiled, not run on hardware
//
#include <hip/hip_runtime.h>

// ---------------------------------------------------------------------------
// CDNA5 (gfx1250) GhostNet-YOLO backbone, f16 NHWC activations, BN folded,
// all 1x1 convs + 3x3/s2 downsamples via v_wmma_f32_16x16x32_f16.
// Tile staging uses GLOBAL_LOAD_ASYNC_TO_LDS_B128 (ASYNCcnt) when available.
// ---------------------------------------------------------------------------

typedef _Float16 v8h  __attribute__((ext_vector_type(8)));
typedef _Float16 v16h __attribute__((ext_vector_type(16)));
typedef float    v8f  __attribute__((ext_vector_type(8)));

union HFrag { v16h v; v8h h[2]; };

#if __has_builtin(__builtin_amdgcn_global_load_async_to_lds_b128)
#define HAVE_ASYNC_LDS 1
#else
#define HAVE_ASYNC_LDS 0
#endif

#if HAVE_ASYNC_LDS
// Param 1: AS(1) pointer to int __attribute__((vector_size(16)))  (global src)
// Param 2: AS(3) pointer to same vector type                      (LDS dst)
// Param 3: imm offset (applies to both addresses), Param 4: imm cpol.
typedef int v4i __attribute__((vector_size(16)));
typedef __attribute__((address_space(1))) v4i gv4i;
typedef __attribute__((address_space(3))) v4i lv4i;
#define ASYNC_CP_B128(dst, src, off)                                   \
  __builtin_amdgcn_global_load_async_to_lds_b128(                      \
      (gv4i*)(void*)(src), (lv4i*)(void*)(dst), (off), 0)
#endif

__device__ __forceinline__ void async_wait_all() {
#if HAVE_ASYNC_LDS
#if __has_builtin(__builtin_amdgcn_s_wait_asynccnt)
  __builtin_amdgcn_s_wait_asynccnt(0);
#else
  asm volatile("s_wait_asynccnt 0" ::: "memory");
#endif
#endif
}

__device__ __forceinline__ float silu_f(float x) { return x / (1.0f + __expf(-x)); }

// ---------------------------------------------------------------------------
// Weight prep: OIHW fp32 + BN -> f16 [O][(kh*K+kw)*I + i] * (g*rsqrt(v+eps)),
// bias[o] = b - m*scale.  (Also used for depthwise with I==1.)
// ---------------------------------------------------------------------------
__global__ void prep_w_kernel(const float* __restrict__ w, const float* __restrict__ g,
                              const float* __restrict__ b, const float* __restrict__ m,
                              const float* __restrict__ v, _Float16* __restrict__ wp,
                              float* __restrict__ bp, int O, int I, int KK) {
  long long t = (long long)blockIdx.x * 256 + threadIdx.x;
  if (t >= (long long)O * I * KK) return;
  int o  = (int)(t / (I * KK));
  int r  = (int)(t % (I * KK));
  int i  = r / KK;
  int kk = r % KK;
  float scale = g[o] * rsqrtf(v[o] + 1e-5f);
  wp[(long long)o * I * KK + (long long)kk * I + i] = (_Float16)(w[t] * scale);
  if (r == 0) bp[o] = b[o] - m[o] * scale;
}

// Stem weights kept fp32, relaid to [(tap*3+ci)*32 + co] for broadcast use.
__global__ void prep_stem_kernel(const float* __restrict__ w, const float* __restrict__ g,
                                 const float* __restrict__ b, const float* __restrict__ m,
                                 const float* __restrict__ v, float* __restrict__ wp,
                                 float* __restrict__ bp) {
  int t = blockIdx.x * 256 + threadIdx.x;
  if (t >= 32 * 108) return;
  int o  = t / 108;
  int r  = t % 108;
  int ci = r / 36;
  int kk = r % 36;
  float scale = g[o] * rsqrtf(v[o] + 1e-5f);
  wp[(kk * 3 + ci) * 32 + o] = w[t] * scale;
  if (r == 0) bp[o] = b[o] - m[o] * scale;
}

// ---------------------------------------------------------------------------
// Stem: 6x6 s2 p2 conv, 3->32, NCHW fp32 in -> NHWC f16 out (+BN+SiLU).
// ---------------------------------------------------------------------------
__global__ __launch_bounds__(128) void stem_kernel(const float* __restrict__ Xin,
                                                   const float* __restrict__ Wst,
                                                   const float* __restrict__ Bst,
                                                   _Float16* __restrict__ Y) {
  __shared__ float sW[108 * 32];
  for (int i = threadIdx.x; i < 108 * 32; i += 128) sW[i] = Wst[i];
  __syncthreads();
  long long col = (long long)blockIdx.x * 128 + threadIdx.x;  // 8*256*256 cols
  int n  = (int)(col >> 16);
  int rm = (int)(col & 65535);
  int oh = rm >> 8, ow = rm & 255;
  float acc[32];
#pragma unroll
  for (int i = 0; i < 32; i++) acc[i] = 0.0f;
  for (int kh = 0; kh < 6; kh++) {
    int ih = oh * 2 + kh - 2;
    if ((unsigned)ih >= 512u) continue;
    for (int kw = 0; kw < 6; kw++) {
      int iw = ow * 2 + kw - 2;
      if ((unsigned)iw >= 512u) continue;
      int tap = kh * 6 + kw;
#pragma unroll
      for (int ci = 0; ci < 3; ci++) {
        float xv = Xin[(((long long)n * 3 + ci) * 512 + ih) * 512 + iw];
        const float* wr = &sW[(tap * 3 + ci) * 32];
#pragma unroll
        for (int co = 0; co < 32; co++) acc[co] += xv * wr[co];
      }
    }
  }
  _Float16* yp = Y + col * 32;
#pragma unroll
  for (int co = 0; co < 32; co++) yp[co] = (_Float16)silu_f(acc[co] + Bst[co]);
}

// ---------------------------------------------------------------------------
// Implicit-GEMM conv (1x1 or KHxKW/stride/pad) with WMMA f16 -> f32.
//   D[m, col] = sum_{tap,k} W[m][tap*Kt+k] * X[pix(col,tap)][cinOff+k]
// Block tile 32M x 128N x 32K, 4 waves each 16x64, double-buffered LDS.
// Staging via async global->LDS b128; epilogue +bias, SiLU, f16 NHWC store.
// ---------------------------------------------------------------------------
#define BM 32
#define BN 128
#define BK 32

template <bool IS1X1>
__global__ __launch_bounds__(128) void gemm_conv_kernel(
    const _Float16* __restrict__ W, const _Float16* __restrict__ X,
    const float* __restrict__ bias, _Float16* __restrict__ Y,
    int Mtot, int Kt, int taps, long long cols,
    int ldx, int cinOff, int ldy, int coutOff,
    int KW, int stride, int pad, int Hin, int Win, int Ho, int Wo, int flags) {
  __shared__ _Float16 sA[2][BM * BK];
  __shared__ _Float16 sB[2][BN * BK];

  const int tid    = threadIdx.x;
  const int lane   = tid & 31;
  const int wave   = tid >> 5;
  const int wm     = wave & 1;          // M half (0..1)
  const int wn     = wave >> 1;         // N half (0..1)
  const int laneN  = lane & 15;
  const int laneHi = lane >> 4;

  const int m0         = blockIdx.y * BM;
  const long long col0 = (long long)blockIdx.x * BN;
  const int ldw        = taps * Kt;
  const int arow       = tid >> 2, aseg = tid & 3;

  // Per-thread B-loader geometry (col fixed for whole K loop).
  long long bcol = col0 + tid;
  bool colOK = bcol < cols;
  int bn_ = 0, boh = 0, bow = 0;
  if (!IS1X1) {
    long long hw = (long long)Ho * Wo;
    long long c  = colOK ? bcol : 0;
    bn_ = (int)(c / hw);
    long long r = c - (long long)bn_ * hw;
    boh = (int)(r / Wo);
    bow = (int)(r % Wo);
  }

  const int kSteps    = Kt / BK;
  const int totalIter = taps * kSteps;

  auto loadAB = [&](int it, int buf) {
    int tap = IS1X1 ? 0 : (it / kSteps);
    int k0  = (IS1X1 ? it : (it % kSteps)) * BK;
    // --- A tile: 32 rows x 32 halfs, row-major [M][K]; 16B per thread
    {
      int mm = m0 + arow;
      v8h* adst = (v8h*)&sA[buf][arow * BK + aseg * 8];
      const _Float16* asrc = W + (size_t)mm * ldw + (size_t)tap * Kt + k0 + aseg * 8;
      if (mm < Mtot) {
#if HAVE_ASYNC_LDS
        ASYNC_CP_B128(adst, asrc, 0);
#else
        *adst = *(const v8h*)asrc;
#endif
      } else {
        v8h z = {};
        *adst = z;
      }
    }
    // --- B tile: this thread's col -> 32 halfs (64B), stored [N][K]
    {
      bool ok = colOK;
      long long scol = bcol;
      if (!IS1X1) {
        int kh = tap / KW, kw = tap - kh * KW;
        int ih = boh * stride + kh - pad;
        int iw = bow * stride + kw - pad;
        ok   = colOK && ((unsigned)ih < (unsigned)Hin) && ((unsigned)iw < (unsigned)Win);
        scol = ((long long)bn_ * Hin + ih) * Win + iw;
      }
      v8h* dst = (v8h*)&sB[buf][tid * BK];
      if (ok) {
        const _Float16* src = X + scol * (long long)ldx + cinOff + k0;
#if HAVE_ASYNC_LDS
        ASYNC_CP_B128(dst, src, 0);
        ASYNC_CP_B128(dst, src, 16);
        ASYNC_CP_B128(dst, src, 32);
        ASYNC_CP_B128(dst, src, 48);
#else
        const v8h* sv = (const v8h*)src;
        v8h vv0 = sv[0], vv1 = sv[1], vv2 = sv[2], vv3 = sv[3];
        __builtin_prefetch(src + BK, 0, 1);  // global_prefetch_b8
        dst[0] = vv0; dst[1] = vv1; dst[2] = vv2; dst[3] = vv3;
#endif
      } else {
        v8h z = {};
        dst[0] = z; dst[1] = z; dst[2] = z; dst[3] = z;
      }
    }
  };

  v8f acc[4] = {};
  loadAB(0, 0);
  for (int it = 0; it < totalIter; ++it) {
    async_wait_all();     // this wave's ASYNCcnt -> 0 before publishing tile
    __syncthreads();      // compiler adds dscnt wait for the zero-fill stores
    if (it + 1 < totalIter) loadAB(it + 1, (it + 1) & 1);
    const int buf = it & 1;

    // Load A fragment + all 4 B fragments first so the DS reads batch into
    // one clause with a single s_wait_dscnt before the 4-WMMA chain.
    HFrag a;
    const _Float16* Ar = &sA[buf][(wm * 16 + laneN) * BK];
    a.h[0] = *(const v8h*)(Ar + laneHi * 16);
    a.h[1] = *(const v8h*)(Ar + laneHi * 16 + 8);

    const _Float16* Bbase = &sB[buf][(wn * 64 + laneN) * BK + laneHi * 16];
    HFrag b0, b1, b2, b3;
    b0.h[0] = *(const v8h*)(Bbase + 0 * 16 * BK);
    b0.h[1] = *(const v8h*)(Bbase + 0 * 16 * BK + 8);
    b1.h[0] = *(const v8h*)(Bbase + 1 * 16 * BK);
    b1.h[1] = *(const v8h*)(Bbase + 1 * 16 * BK + 8);
    b2.h[0] = *(const v8h*)(Bbase + 2 * 16 * BK);
    b2.h[1] = *(const v8h*)(Bbase + 2 * 16 * BK + 8);
    b3.h[0] = *(const v8h*)(Bbase + 3 * 16 * BK);
    b3.h[1] = *(const v8h*)(Bbase + 3 * 16 * BK + 8);

    acc[0] = __builtin_amdgcn_wmma_f32_16x16x32_f16(false, a.v, false, b0.v, (short)0, acc[0], false, false);
    acc[1] = __builtin_amdgcn_wmma_f32_16x16x32_f16(false, a.v, false, b1.v, (short)0, acc[1], false, false);
    acc[2] = __builtin_amdgcn_wmma_f32_16x16x32_f16(false, a.v, false, b2.v, (short)0, acc[2], false, false);
    acc[3] = __builtin_amdgcn_wmma_f32_16x16x32_f16(false, a.v, false, b3.v, (short)0, acc[3], false, false);
  }

  // Epilogue: bias + SiLU + f16 pack, 8 consecutive channels per lane.
  const int mBase = m0 + wm * 16 + laneHi * 8;
  if (mBase < Mtot) {
#pragma unroll
    for (int s = 0; s < 4; s++) {
      long long ccol = col0 + wn * 64 + s * 16 + laneN;
      if (ccol < cols) {
        v8h ov;
#pragma unroll
        for (int r = 0; r < 8; r++) {
          float vv = acc[s][r] + bias[mBase + r];
          if (flags & 1) vv = silu_f(vv);
          ov[r] = (_Float16)vv;
        }
        *(v8h*)(Y + ccol * (long long)ldy + coutOff + mBase) = ov;
      }
    }
  }
}

// ---------------------------------------------------------------------------
// Depthwise 3x3 s1 p1 + BN + SiLU (bandwidth bound).
// ---------------------------------------------------------------------------
__global__ void dwconv3x3_kernel(const _Float16* __restrict__ X, const _Float16* __restrict__ Wd,
                                 const float* __restrict__ bias, _Float16* __restrict__ Y,
                                 int C, int ldx, int cinOff, int ldy, int coutOff,
                                 int H, int Wimg, long long cols) {
  long long t = (long long)blockIdx.x * 256 + threadIdx.x;
  if (t >= cols * C) return;
  int c = (int)(t % C);
  long long col = t / C;
  long long hw = (long long)H * Wimg;
  int n = (int)(col / hw);
  long long r = col - (long long)n * hw;
  int h = (int)(r / Wimg), w = (int)(r % Wimg);
  float acc = bias[c];
#pragma unroll
  for (int kh = 0; kh < 3; kh++) {
    int ih = h + kh - 1;
    if ((unsigned)ih >= (unsigned)H) continue;
#pragma unroll
    for (int kw = 0; kw < 3; kw++) {
      int iw = w + kw - 1;
      if ((unsigned)iw >= (unsigned)Wimg) continue;
      acc += (float)X[(((long long)n * H + ih) * Wimg + iw) * ldx + cinOff + c] *
             (float)Wd[c * 9 + kh * 3 + kw];
    }
  }
  Y[col * (long long)ldy + coutOff + c] = (_Float16)silu_f(acc);
}

// ---------------------------------------------------------------------------
// Elementwise: in-place SiLU; residual add + SiLU (into strided concat slot).
// ---------------------------------------------------------------------------
__global__ void ew_silu_kernel(_Float16* __restrict__ X, long long nElem) {
  long long t = (long long)blockIdx.x * 256 + threadIdx.x;
  if (t >= nElem) return;
  X[t] = (_Float16)silu_f((float)X[t]);
}

__global__ void ew_add_silu_kernel(const _Float16* __restrict__ A, const _Float16* __restrict__ R,
                                   _Float16* __restrict__ Out, int C, int ldo, int ooff,
                                   long long cols) {
  long long t = (long long)blockIdx.x * 256 + threadIdx.x;
  if (t >= cols * C) return;
  int c = (int)(t % C);
  long long col = t / C;
  float v = (float)A[t] + (float)R[t];
  Out[col * (long long)ldo + ooff + c] = (_Float16)silu_f(v);
}

// ---------------------------------------------------------------------------
// SimAM: per (n,c) single-pass sum / sumsq reduction then pointwise gate.
// ---------------------------------------------------------------------------
__global__ void simam_kernel(const _Float16* __restrict__ X, _Float16* __restrict__ Y,
                             int C, int HW) {
  __shared__ float s1[256], s2[256];
  int b = blockIdx.x;
  int n = b / C, c = b % C;
  const _Float16* base = X + (long long)n * HW * C + c;
  float sum = 0.0f, sq = 0.0f;
  for (int p = threadIdx.x; p < HW; p += 256) {
    float v = (float)base[(long long)p * C];
    sum += v; sq += v * v;
  }
  s1[threadIdx.x] = sum; s2[threadIdx.x] = sq;
  __syncthreads();
  for (int st = 128; st > 0; st >>= 1) {
    if (threadIdx.x < st) { s1[threadIdx.x] += s1[threadIdx.x + st]; s2[threadIdx.x] += s2[threadIdx.x + st]; }
    __syncthreads();
  }
  float mu    = s1[0] / (float)HW;
  float sumsq = s2[0] - (float)HW * mu * mu;          // sum of (x-mu)^2
  float denom = 4.0f * (sumsq / (float)(HW - 1) + 1e-4f);
  _Float16* ob = Y + (long long)n * HW * C + c;
  for (int p = threadIdx.x; p < HW; p += 256) {
    float v = (float)base[(long long)p * C];
    float d = (v - mu) * (v - mu);
    float yv = d / denom + 0.5f;
    ob[(long long)p * C] = (_Float16)(v * (1.0f / (1.0f + __expf(-yv))));
  }
}

// ---------------------------------------------------------------------------
// Head: 1x1 conv 512->45, f16 NHWC in, fp32 NCHW out (d_out). Tiny GEMM.
// ---------------------------------------------------------------------------
__global__ void head_kernel(const _Float16* __restrict__ F, const float* __restrict__ Wt,
                            const float* __restrict__ Bb, float* __restrict__ Out,
                            long long cols, int HW) {
  long long t = (long long)blockIdx.x * 256 + threadIdx.x;
  int co = (int)(t & 63);
  long long col = t >> 6;
  if (co >= 45 || col >= cols) return;
  const _Float16* f = F + col * 512;
  const float* w = Wt + (long long)co * 512;
  float acc = Bb[co];
  for (int c = 0; c < 512; c++) acc += (float)f[c] * w[c];
  int n = (int)(col / HW), s = (int)(col % HW);
  Out[((long long)n * 45 + co) * HW + s] = acc;
}

// ---------------------------------------------------------------------------
// Host orchestration
// ---------------------------------------------------------------------------
struct ConvBN { const float *b, *g, *m, *v, *w; int cout, cin, k; };
struct PrepW  { _Float16* wp; float* bp; };
struct Ghost2 { ConvBN cheap, primary; PrepW pc, pp; };
struct Bneck  { Ghost2 c1, c3; };
struct C3Blk  { ConvBN conv1, conv2, conv3; PrepW p1, p2, p3; Bneck m[3]; int n, h, cin, cout; };

extern "C" void kernel_launch(void* const* d_in, const int* in_sizes, int n_in,
                              void* d_out, int out_size, void* d_ws, size_t ws_size,
                              hipStream_t stream) {
  (void)in_sizes; (void)n_in; (void)out_size; (void)ws_size;

  // --- input walker (jax pytree order: sorted dict keys, lists in order) ---
  int ix = 1;  // d_in[0] == x
  auto nextConv = [&](int cout, int cin, int k) {
    ConvBN c;
    c.b = (const float*)d_in[ix++]; c.g = (const float*)d_in[ix++];
    c.m = (const float*)d_in[ix++]; c.v = (const float*)d_in[ix++];
    c.w = (const float*)d_in[ix++];
    c.cout = cout; c.cin = cin; c.k = k; return c;
  };
  auto nextGhost2 = [&](int cin, int cout) {  // keys: cheap < primary
    Ghost2 g;
    int init = cout / 2;
    g.cheap   = nextConv(init, 1, 3);      // depthwise
    g.primary = nextConv(init, cin, 1);
    return g;
  };
  auto nextC3 = [&](int cin, int cout, int n) {  // keys: conv1,conv2,conv3,m
    C3Blk B; B.n = n; B.cin = cin; B.cout = cout; B.h = cout / 2;
    B.conv1 = nextConv(B.h, cin, 1);
    B.conv2 = nextConv(B.h, cin, 1);
    B.conv3 = nextConv(cout, 2 * B.h, 1);
    for (int i = 0; i < n; i++) { B.m[i].c1 = nextGhost2(B.h, B.h); B.m[i].c3 = nextGhost2(B.h, B.h); }
    return B;
  };

  C3Blk c31 = nextC3(32, 64, 1);
  C3Blk c32 = nextC3(128, 128, 3);
  C3Blk c33 = nextC3(256, 256, 3);
  C3Blk c34 = nextC3(512, 512, 1);
  ConvBN down1 = nextConv(128, 64, 3);
  ConvBN down2 = nextConv(256, 128, 3);
  ConvBN down3 = nextConv(512, 256, 3);
  const float* headB[3]; const float* headW[3];
  for (int h = 0; h < 3; h++) { headB[h] = (const float*)d_in[ix++]; headW[h] = (const float*)d_in[ix++]; }
  ConvBN stem = nextConv(32, 3, 6);

  // --- workspace bump allocator ---
  size_t off = 0;
  auto alloc = [&](size_t bytes) {
    size_t o = (off + 255) & ~(size_t)255;
    off = o + bytes;
    return (void*)((char*)d_ws + o);
  };
  auto prep = [&](const ConvBN& c) {
    int KK = c.k * c.k, I = c.cin;
    long long total = (long long)c.cout * I * KK;
    PrepW p;
    p.wp = (_Float16*)alloc((size_t)total * sizeof(_Float16));
    p.bp = (float*)alloc((size_t)c.cout * sizeof(float));
    prep_w_kernel<<<(unsigned)((total + 255) / 256), 256, 0, stream>>>(
        c.w, c.g, c.b, c.m, c.v, p.wp, p.bp, c.cout, I, KK);
    return p;
  };
  auto prepC3 = [&](C3Blk& B) {
    B.p1 = prep(B.conv1); B.p2 = prep(B.conv2); B.p3 = prep(B.conv3);
    for (int i = 0; i < B.n; i++) {
      B.m[i].c1.pc = prep(B.m[i].c1.cheap); B.m[i].c1.pp = prep(B.m[i].c1.primary);
      B.m[i].c3.pc = prep(B.m[i].c3.cheap); B.m[i].c3.pp = prep(B.m[i].c3.primary);
    }
  };
  prepC3(c31); prepC3(c32); prepC3(c33); prepC3(c34);
  PrepW pd1 = prep(down1), pd2 = prep(down2), pd3 = prep(down3);
  float* wstem = (float*)alloc(108 * 32 * sizeof(float));
  float* bstem = (float*)alloc(32 * sizeof(float));
  prep_stem_kernel<<<(32 * 108 + 255) / 256, 256, 0, stream>>>(
      stem.w, stem.g, stem.b, stem.m, stem.v, wstem, bstem);

  // --- activation buffers (f16 NHWC) ---
  const size_t EL_MAIN = 33554432;  // 64ch * 524288 cols (largest tensor)
  const size_t EL_HALF = 16777216;
  _Float16* MAIN0 = (_Float16*)alloc(EL_MAIN * 2);
  _Float16* MAIN1 = (_Float16*)alloc(EL_MAIN * 2);
  _Float16* CAT   = (_Float16*)alloc(EL_MAIN * 2);
  _Float16* X1b   = (_Float16*)alloc(EL_HALF * 2);
  _Float16* G1    = (_Float16*)alloc(EL_HALF * 2);
  _Float16* G2    = (_Float16*)alloc(EL_HALF * 2);
  _Float16* Tb    = (_Float16*)alloc(EL_HALF * 2);

  // --- launch helpers ---
  auto gemm1x1 = [&](const _Float16* X, int ldx, int Kt, const PrepW& w, int Mtot,
                     _Float16* Y, int ldy, int coutOff, long long cols) {
    dim3 g((unsigned)((cols + 127) / 128), (unsigned)((Mtot + 31) / 32));
    gemm_conv_kernel<true><<<g, 128, 0, stream>>>(
        w.wp, X, w.bp, Y, Mtot, Kt, 1, cols, ldx, 0, ldy, coutOff,
        1, 1, 0, 0, 0, 0, 0, 1);
  };
  auto conv3s2 = [&](const _Float16* X, int cin, int Hin, int Win, const PrepW& w,
                     int cout, _Float16* Y) {
    int Ho = Hin / 2, Wo = Win / 2;
    long long colsO = 8LL * Ho * Wo;
    dim3 g((unsigned)((colsO + 127) / 128), (unsigned)((cout + 31) / 32));
    gemm_conv_kernel<false><<<g, 128, 0, stream>>>(
        w.wp, X, w.bp, Y, cout, cin, 9, colsO, cin, 0, cout, 0,
        3, 2, 1, Hin, Win, Ho, Wo, 1);
  };
  auto dwc = [&](const _Float16* X, int C, int ldx, int cinOff, _Float16* Y, int ldy,
                 int coutOff, const PrepW& w, int H, int Wimg, long long cols) {
    long long tot = cols * C;
    dwconv3x3_kernel<<<(unsigned)((tot + 255) / 256), 256, 0, stream>>>(
        X, w.wp, w.bp, Y, C, ldx, cinOff, ldy, coutOff, H, Wimg, cols);
  };
  auto run_c3 = [&](C3Blk& B, const _Float16* in, int cin, long long cols, int H, int Wimg,
                    _Float16* out) {
    int h = B.h, cout = B.cout;
    gemm1x1(in, cin, cin, B.p1, h, X1b, h, 0, cols);        // x1
    gemm1x1(in, cin, cin, B.p2, h, CAT, 2 * h, h, cols);    // x2 -> CAT[h:2h]
    const _Float16* cur = X1b;
    _Float16* spare = Tb;
    for (int i = 0; i < B.n; i++) {
      Bneck& bn = B.m[i];
      // c1 ghostconv -> G1 (primary [0:h/2], dw [h/2:h]), then extra SiLU
      gemm1x1(cur, h, h, bn.c1.pp, h / 2, G1, h, 0, cols);
      dwc(G1, h / 2, h, 0, G1, h, h / 2, bn.c1.pc, H, Wimg, cols);
      ew_silu_kernel<<<(unsigned)((cols * h + 255) / 256), 256, 0, stream>>>(G1, cols * (long long)h);
      // c3 ghostconv -> G2
      gemm1x1(G1, h, h, bn.c3.pp, h / 2, G2, h, 0, cols);
      dwc(G2, h / 2, h, 0, G2, h, h / 2, bn.c3.pc, H, Wimg, cols);
      // x1 = silu(G2 + cur); last iteration writes straight into CAT[0:h]
      long long tot = cols * h;
      if (i == B.n - 1) {
        ew_add_silu_kernel<<<(unsigned)((tot + 255) / 256), 256, 0, stream>>>(
            G2, cur, CAT, h, 2 * h, 0, cols);
      } else {
        ew_add_silu_kernel<<<(unsigned)((tot + 255) / 256), 256, 0, stream>>>(
            G2, cur, spare, h, h, 0, cols);
        const _Float16* t = cur; cur = spare; spare = (_Float16*)t;
      }
    }
    gemm1x1(CAT, 2 * h, 2 * h, B.p3, cout, out, cout, 0, cols);
  };

  // --- network ---
  const long long COLS0 = 8LL * 256 * 256;  // after stem
  stem_kernel<<<(unsigned)(COLS0 / 128), 128, 0, stream>>>(
      (const float*)d_in[0], wstem, bstem, MAIN0);

  run_c3(c31, MAIN0, 32, COLS0, 256, 256, MAIN1);          // 64ch @256
  conv3s2(MAIN1, 64, 256, 256, pd1, 128, MAIN0);           // 128ch @128
  run_c3(c32, MAIN0, 128, 8LL * 128 * 128, 128, 128, MAIN1);
  conv3s2(MAIN1, 128, 128, 128, pd2, 256, MAIN0);          // 256ch @64
  run_c3(c33, MAIN0, 256, 8LL * 64 * 64, 64, 64, MAIN1);
  conv3s2(MAIN1, 256, 64, 64, pd3, 512, MAIN0);            // 512ch @32
  run_c3(c34, MAIN0, 512, 8LL * 32 * 32, 32, 32, MAIN1);

  simam_kernel<<<8 * 512, 256, 0, stream>>>(MAIN1, MAIN0, 512, 1024);

  const long long HCOLS = 8LL * 32 * 32;  // 8192
  for (int h = 0; h < 3; h++) {
    head_kernel<<<(unsigned)((HCOLS * 64 + 255) / 256), 256, 0, stream>>>(
        MAIN0, headW[h], headB[h], (float*)d_out + (long long)h * 8 * 45 * 1024,
        HCOLS, 1024);
  }
}
